// PointSAM_3Headed_87273735455099
// MI455X (gfx1250) — matrix operand order repaired and checked by reading, not compiled, not run on hardware
//
#include <hip/hip_runtime.h>
#include <math.h>

// ---------------------------------------------------------------------------
// PointSAM forward for gfx1250 (MI455X).  All MLP GEMMs run on
// v_wmma_f32_16x16x32_f16 (f16 A/B, f32 accum) with 32x32 register blocking
// per wave; weight tiles are double-buffered through LDS with
// global_load_async_to_lds_b128 (ASYNCcnt) overlapped with WMMA compute.
// BN is folded into per-channel alpha/beta.  Everything else (kNN, grouping,
// maxpool, 3-NN interpolation, IoU negative-prompt logic) is scalar/LDS.
// ---------------------------------------------------------------------------

typedef __attribute__((ext_vector_type(16))) _Float16 v16h;
typedef __attribute__((ext_vector_type(8)))  _Float16 v8h;
typedef __attribute__((ext_vector_type(8)))  float    v8f;

#define BN_EPS 1e-5f

// ---- CDNA5 async global->LDS copy (tracked by ASYNCcnt) --------------------
__device__ __forceinline__ void async_b128_to_lds(unsigned lds_off, const _Float16* g)
{
    asm volatile("global_load_async_to_lds_b128 %0, %1, off"
                 :: "v"(lds_off), "v"(g)
                 : "memory");
}
__device__ __forceinline__ void wait_async_prev()   // allow 2 newest in flight
{
    asm volatile("s_wait_asynccnt 0x2" ::: "memory");
}
__device__ __forceinline__ void wait_async_all()
{
    asm volatile("s_wait_asynccnt 0x0" ::: "memory");
}

// ============================ WMMA GEMM =====================================
// Y[m,n] = relu( al[n] * sum_k X[m,k]*W[n,k] + be[n] )
// X: M x Kpad f16 row-major (Kpad % 32 == 0, pad = 0)
// W: Cout x Kpad f16 row-major (B^T; contiguous along K)
// Yh: M x Cout f16 (always written), Yf: optional f32 mirror.
// Block = 128 threads = 4 waves covering a 64(M) x 64(N) macro-tile; wave w
// owns rows [blk.x*64 + 32*(w&1), +32) x cols [blk.y*64 + 32*(w>>1), +32):
// 2x2 WMMA tiles, 4 accumulators.  B slab (64 cols x 32 K, 4KB) staged in LDS
// via async b128 copies, double buffered against the WMMA loop.
__global__ void __launch_bounds__(128)
gemm_bn_relu_wmma(const _Float16* __restrict__ X,
                  const _Float16* __restrict__ W,
                  const float*    __restrict__ al,
                  const float*    __restrict__ be,
                  _Float16*       __restrict__ Yh,
                  float*          __restrict__ Yf,
                  int M, int Kpad, int Cout)
{
    __shared__ __align__(64) _Float16 smB[2][64 * 32];   // 2 x 4KB

    const int tid  = threadIdx.x;
    const int lane = tid & 31;
    const int wave = tid >> 5;
    const int hlf  = lane >> 4;       // 16-lane half of the wave
    const int l16  = lane & 15;

    const int mBase = blockIdx.x * 64 + (wave & 1) * 32;
    const int nBlk  = blockIdx.y * 64;
    const int nSub  = (wave >> 1) * 32;

    // --- staging assignment: 64 rows x 4 b128-segments = 256 = 128 thr x 2 --
    const int q0 = tid * 2;
    const int j0 = q0 >> 2,       s0 = (q0 & 3) * 8;          // row, half-off
    const int j1 = (q0 + 1) >> 2, s1 = ((q0 + 1) & 3) * 8;
    const _Float16* Wg0 = W + (size_t)(nBlk + j0) * Kpad + s0;
    const _Float16* Wg1 = W + (size_t)(nBlk + j1) * Kpad + s1;
    const unsigned  lb0 = (unsigned)(size_t)&smB[0][j0 * 32 + s0];
    const unsigned  lb1 = (unsigned)(size_t)&smB[0][j1 * 32 + s1];
    const unsigned  bufStride = (unsigned)(64 * 32 * sizeof(_Float16)); // 4096B

    const _Float16* __restrict__ Xr0 = X + (size_t)(mBase + l16) * Kpad;
    const _Float16* __restrict__ Xr1 = Xr0 + (size_t)16 * Kpad;

    v8f acc00 = {0.f, 0.f, 0.f, 0.f, 0.f, 0.f, 0.f, 0.f};
    v8f acc01 = acc00, acc10 = acc00, acc11 = acc00;

    // prologue: stage K-chunk 0 into buffer 0
    async_b128_to_lds(lb0, Wg0);
    async_b128_to_lds(lb1, Wg1);

    int cur = 0;
    for (int kt = 0; kt < Kpad; kt += 32) {
        const int nxt = kt + 32;
        if (nxt < Kpad) {
            // stage next chunk into the other buffer while we compute
            async_b128_to_lds(lb0 + (unsigned)(cur ^ 1) * bufStride, Wg0 + nxt);
            async_b128_to_lds(lb1 + (unsigned)(cur ^ 1) * bufStride, Wg1 + nxt);
            wait_async_prev();                 // current chunk's copies done
            __builtin_prefetch(Xr0 + nxt, 0, 3);
            __builtin_prefetch(Xr1 + nxt, 0, 3);
        } else {
            wait_async_all();
        }
        __syncthreads();

        // A fragments (16x32 f16): halves 0..7 <- K = kt+16*hlf .. +7,
        //                          halves 8..15 <- K = kt+16*hlf+8 .. +7
        v8h a00 = *(const v8h*)(Xr0 + kt + 16 * hlf);
        v8h a01 = *(const v8h*)(Xr0 + kt + 16 * hlf + 8);
        v8h a10 = *(const v8h*)(Xr1 + kt + 16 * hlf);
        v8h a11 = *(const v8h*)(Xr1 + kt + 16 * hlf + 8);
        v16h A0, A1;
#pragma unroll
        for (int i = 0; i < 8; ++i) {
            A0[i] = a00[i]; A0[i + 8] = a01[i];
            A1[i] = a10[i]; A1[i + 8] = a11[i];
        }
        // B fragments (32x16 f16) from LDS: lane = col, 16 contig K at 16*hlf
        const _Float16* Bp = &smB[cur][0];
        v16h B0 = *(const v16h*)(Bp + (nSub + l16) * 32 + 16 * hlf);
        v16h B1 = *(const v16h*)(Bp + (nSub + 16 + l16) * 32 + 16 * hlf);

        acc00 = __builtin_amdgcn_wmma_f32_16x16x32_f16(false, A0, false, B0, (short)0, acc00, false, false);
        acc01 = __builtin_amdgcn_wmma_f32_16x16x32_f16(false, A0, false, B1, (short)0, acc01, false, false);
        acc10 = __builtin_amdgcn_wmma_f32_16x16x32_f16(false, A1, false, B0, (short)0, acc10, false, false);
        acc11 = __builtin_amdgcn_wmma_f32_16x16x32_f16(false, A1, false, B1, (short)0, acc11, false, false);

        __syncthreads();   // everyone done reading smB[cur] before overwrite
        cur ^= 1;
    }

    const int   n0  = nBlk + nSub + l16;
    const int   n1  = n0 + 16;
    const float a0s = al[n0], b0s = be[n0];
    const float a1s = al[n1], b1s = be[n1];
#pragma unroll
    for (int r = 0; r < 8; ++r) {
        const int m0 = mBase + r + 8 * hlf;   // C/D row mapping
        const int m1 = m0 + 16;
        float y;
        y = a0s * acc00[r] + b0s; y = y > 0.f ? y : 0.f;
        Yh[(size_t)m0 * Cout + n0] = (_Float16)y; if (Yf) Yf[(size_t)m0 * Cout + n0] = y;
        y = a1s * acc01[r] + b1s; y = y > 0.f ? y : 0.f;
        Yh[(size_t)m0 * Cout + n1] = (_Float16)y; if (Yf) Yf[(size_t)m0 * Cout + n1] = y;
        y = a0s * acc10[r] + b0s; y = y > 0.f ? y : 0.f;
        Yh[(size_t)m1 * Cout + n0] = (_Float16)y; if (Yf) Yf[(size_t)m1 * Cout + n0] = y;
        y = a1s * acc11[r] + b1s; y = y > 0.f ? y : 0.f;
        Yh[(size_t)m1 * Cout + n1] = (_Float16)y; if (Yf) Yf[(size_t)m1 * Cout + n1] = y;
    }
}

// ====================== layer prep: fold BN, cast W to f16 ==================
__global__ void prep_layer(const float* __restrict__ W, const float* __restrict__ bias,
                           const float* __restrict__ gamma, const float* __restrict__ beta,
                           const float* __restrict__ mean,  const float* __restrict__ var,
                           _Float16* __restrict__ Wh, float* __restrict__ al,
                           float* __restrict__ be, int Cout, int Cin, int Kpad)
{
    int t = blockIdx.x * blockDim.x + threadIdx.x;
    if (t >= Cout * Kpad) return;
    int o = t / Kpad, k = t % Kpad;
    Wh[t] = (_Float16)(k < Cin ? W[(size_t)o * Cin + k] : 0.f);
    if (k == 0) {
        if (gamma) {
            float a = gamma[o] * rsqrtf(var[o] + BN_EPS);
            al[o] = a;
            be[o] = (bias[o] - mean[o]) * a + beta[o];
        } else {
            al[o] = 1.f;
            be[o] = bias[o];
        }
    }
}

// ============================ kNN top-k =====================================
// One block per query. Distances cached in LDS, then ns rounds of argmin.
__global__ void knn_kernel(const float* __restrict__ q, const float* __restrict__ pts,
                           int* __restrict__ idxOut, int Nq, int Np, int ns)
{
    extern __shared__ float sm[];
    float* dist = sm;                 // Np
    float* rval = sm + Np;            // 256
    int*   ridx = (int*)(rval + 256); // 256

    const int bq = blockIdx.x;        // b*Nq + s
    const int b  = bq / Nq;
    const float* Q = q + (size_t)bq * 3;
    const float qx = Q[0], qy = Q[1], qz = Q[2];
    const float* P = pts + (size_t)b * Np * 3;

    for (int p = threadIdx.x; p < Np; p += 256) {
        float dx = P[3 * p] - qx, dy = P[3 * p + 1] - qy, dz = P[3 * p + 2] - qz;
        dist[p] = dx * dx + dy * dy + dz * dz;
    }
    __syncthreads();

    for (int s = 0; s < ns; ++s) {
        float bv = 3.4e38f; int bi = 0;
        for (int p = threadIdx.x; p < Np; p += 256) {
            float d = dist[p];
            if (d < bv) { bv = d; bi = p; }
        }
        rval[threadIdx.x] = bv; ridx[threadIdx.x] = bi;
        __syncthreads();
        for (int st = 128; st > 0; st >>= 1) {
            if (threadIdx.x < st && rval[threadIdx.x + st] < rval[threadIdx.x]) {
                rval[threadIdx.x] = rval[threadIdx.x + st];
                ridx[threadIdx.x] = ridx[threadIdx.x + st];
            }
            __syncthreads();
        }
        if (threadIdx.x == 0) {
            idxOut[(size_t)bq * ns + s] = ridx[0];
            dist[ridx[0]] = 3.4e38f;
        }
        __syncthreads();
    }
}

// ==================== grouping / gather / concat (-> f16) ===================
__global__ void group_points(const float* __restrict__ xyz, const float* __restrict__ newxyz,
                             const float* __restrict__ feat, const int* __restrict__ idx,
                             _Float16* __restrict__ X, int B, int S, int ns,
                             int Np, int Cf, int Kpad)
{
    int r = blockIdx.x * blockDim.x + threadIdx.x;   // (b*S+s)*ns + k
    if (r >= B * S * ns) return;
    int bs = r / ns;
    int b  = bs / S;
    int j  = idx[r];
    _Float16* Xr = X + (size_t)r * Kpad;
    const float* Q = newxyz + (size_t)bs * 3;
    const float* P = xyz + ((size_t)b * Np + j) * 3;
    Xr[0] = (_Float16)(P[0] - Q[0]);
    Xr[1] = (_Float16)(P[1] - Q[1]);
    Xr[2] = (_Float16)(P[2] - Q[2]);
    const float* F = feat + ((size_t)b * Np + j) * Cf;
    for (int c = 0; c < Cf; ++c) Xr[3 + c] = (_Float16)F[c];
    for (int c = 3 + Cf; c < Kpad; ++c) Xr[c] = (_Float16)0.f;
}

// group-all (sa3): relative to point 0 of each batch, concat features.
__global__ void group_all(const float* __restrict__ xyz, const float* __restrict__ feat,
                          _Float16* __restrict__ X, int B, int S, int Cf, int Kpad)
{
    int r = blockIdx.x * blockDim.x + threadIdx.x;   // b*S + s
    if (r >= B * S) return;
    int b = r / S;
    const float* cen = xyz + (size_t)b * S * 3;
    _Float16* Xr = X + (size_t)r * Kpad;
    Xr[0] = (_Float16)(xyz[3 * r]     - cen[0]);
    Xr[1] = (_Float16)(xyz[3 * r + 1] - cen[1]);
    Xr[2] = (_Float16)(xyz[3 * r + 2] - cen[2]);
    const float* F = feat + (size_t)r * Cf;
    for (int c = 0; c < Cf; ++c) Xr[3 + c] = (_Float16)F[c];
    for (int c = 3 + Cf; c < Kpad; ++c) Xr[c] = (_Float16)0.f;
}

// ============================ max-pool over samples =========================
__global__ void maxpool_ns(const _Float16* __restrict__ A, float* __restrict__ out,
                           int BS, int K, int C)
{
    int t = blockIdx.x * blockDim.x + threadIdx.x;   // bs*C + c
    if (t >= BS * C) return;
    int bs = t / C, c = t % C;
    float m = -3.4e38f;
    for (int k = 0; k < K; ++k) {
        float v = (float)A[((size_t)bs * K + k) * C + c];
        if (v > m) m = v;
    }
    out[t] = m;
}

// ================= feature-prop: 3-NN interp + concat (-> f16) ==============
__global__ void fp_build(const float* __restrict__ xyz1, const float* __restrict__ xyz2,
                         const float* __restrict__ p1, const float* __restrict__ p2,
                         _Float16* __restrict__ X, int B, int N, int S,
                         int C1, int C2, int Kpad)
{
    int r = blockIdx.x * blockDim.x + threadIdx.x;   // b*N + n
    if (r >= B * N) return;
    int b = r / N;
    _Float16* Xr = X + (size_t)r * Kpad;
    for (int c = 0; c < C1; ++c) Xr[c] = (_Float16)p1[(size_t)r * C1 + c];
    if (S == 1) {
        const float* F = p2 + (size_t)b * C2;
        for (int c = 0; c < C2; ++c) Xr[C1 + c] = (_Float16)F[c];
    } else {
        const float qx = xyz1[3 * r], qy = xyz1[3 * r + 1], qz = xyz1[3 * r + 2];
        const float* P = xyz2 + (size_t)b * S * 3;
        float d0 = 3.4e38f, d1 = 3.4e38f, d2 = 3.4e38f;
        int   i0 = 0, i1 = 0, i2 = 0;
        for (int s = 0; s < S; ++s) {
            float dx = P[3 * s] - qx, dy = P[3 * s + 1] - qy, dz = P[3 * s + 2] - qz;
            float d = dx * dx + dy * dy + dz * dz;
            if (d < d0)      { d2 = d1; i2 = i1; d1 = d0; i1 = i0; d0 = d; i0 = s; }
            else if (d < d1) { d2 = d1; i2 = i1; d1 = d;  i1 = s; }
            else if (d < d2) { d2 = d;  i2 = s; }
        }
        float w0 = 1.f / (d0 + 1e-8f), w1 = 1.f / (d1 + 1e-8f), w2 = 1.f / (d2 + 1e-8f);
        float wsum = w0 + w1 + w2;
        w0 /= wsum; w1 /= wsum; w2 /= wsum;
        const float* F0 = p2 + ((size_t)b * S + i0) * C2;
        const float* F1 = p2 + ((size_t)b * S + i1) * C2;
        const float* F2 = p2 + ((size_t)b * S + i2) * C2;
        for (int c = 0; c < C2; ++c)
            Xr[C1 + c] = (_Float16)(w0 * F0[c] + w1 * F1[c] + w2 * F2[c]);
    }
    for (int c = C1 + C2; c < Kpad; ++c) Xr[c] = (_Float16)0.f;
}

// ======================= misc elementwise helpers ===========================
__global__ void copy_first(const float* __restrict__ src, float* __restrict__ dst,
                           int B, int Nsrc, int Ndst)
{
    int t = blockIdx.x * blockDim.x + threadIdx.x;
    if (t >= B * Ndst * 3) return;
    int b = t / (Ndst * 3), rem = t % (Ndst * 3);
    dst[t] = src[(size_t)b * Nsrc * 3 + rem];
}

__global__ void cast_pad(const float* __restrict__ src, _Float16* __restrict__ dst,
                         int M, int C, int Kpad)
{
    int t = blockIdx.x * blockDim.x + threadIdx.x;
    if (t >= M * Kpad) return;
    int m = t / Kpad, k = t % Kpad;
    dst[t] = (_Float16)(k < C ? src[(size_t)m * C + k] : 0.f);
}

// x[b,p,n,c] = feats[b,n,c] + pos_e[b,p,c] (- neg_e[b,p,c]) -> f16
__global__ void head_build(const float* __restrict__ feats, const float* __restrict__ pose,
                           const float* __restrict__ nege, _Float16* __restrict__ X,
                           int B, int P, int N, int C)
{
    long long t = (long long)blockIdx.x * blockDim.x + threadIdx.x;
    long long total = (long long)B * P * N * C;
    if (t >= total) return;
    int c = (int)(t % C);
    long long r = t / C;
    int n = (int)(r % N);
    int bp = (int)(r / N);
    int b = bp / P;
    float v = feats[((size_t)b * N + n) * C + c] + pose[(size_t)bp * C + c];
    if (nege) v -= nege[(size_t)bp * C + c];
    X[t] = (_Float16)v;
}

// final 64->1 dense (+optional sigmoid); H is f16 hidden activations.
__global__ void dense_final(const _Float16* __restrict__ H, const float* __restrict__ W,
                            const float* __restrict__ bias, float* __restrict__ out,
                            int M, int C, int do_sig)
{
    int r = blockIdx.x * blockDim.x + threadIdx.x;
    if (r >= M) return;
    float acc = 0.f;
    for (int c = 0; c < C; ++c) acc += (float)H[(size_t)r * C + c] * W[c];
    acc += bias[0];
    if (do_sig) acc = 1.f / (1.f + expf(-acc));
    out[r] = acc;
}

// tiny 3->Cout linear (prompt encoder)
__global__ void linear3(const float* __restrict__ in, const float* __restrict__ W,
                        const float* __restrict__ bias, float* __restrict__ out,
                        int rows, int Cout)
{
    int t = blockIdx.x * blockDim.x + threadIdx.x;
    if (t >= rows * Cout) return;
    int r = t / Cout, o = t % Cout;
    out[t] = in[3 * r] * W[o * 3] + in[3 * r + 1] * W[o * 3 + 1] +
             in[3 * r + 2] * W[o * 3 + 2] + bias[o];
}

// ===================== negative prompt (NPC) logic ==========================
// inter[b,i,j] = sum_n mask_i * mask_j  (mask = logits > 0)
__global__ void npc_inter(const float* __restrict__ logits, float* __restrict__ inter,
                          int B, int P, int N)
{
    __shared__ float red[256];
    int blk = blockIdx.x;                 // b*P*P + i*P + j
    int j = blk % P, i = (blk / P) % P, b = blk / (P * P);
    const float* Li = logits + ((size_t)b * P + i) * N;
    const float* Lj = logits + ((size_t)b * P + j) * N;
    float s = 0.f;
    for (int n = threadIdx.x; n < N; n += 256)
        s += (Li[n] > 0.f && Lj[n] > 0.f) ? 1.f : 0.f;
    red[threadIdx.x] = s;
    __syncthreads();
    for (int st = 128; st > 0; st >>= 1) {
        if (threadIdx.x < st) red[threadIdx.x] += red[threadIdx.x + st];
        __syncthreads();
    }
    if (threadIdx.x == 0) inter[blk] = red[0];
}

__global__ void npc_select(const float* __restrict__ inter, const float* __restrict__ pos,
                           float* __restrict__ neg, int B, int P, float tau)
{
    int t = blockIdx.x * blockDim.x + threadIdx.x;   // b*P + i
    if (t >= B * P) return;
    int b = t / P, i = t % P;
    const float* I = inter + (size_t)b * P * P;
    float si = I[i * P + i];
    int pick = -1;
    for (int j = 0; j < P; ++j) {
        if (j == i) continue;
        float in_ = I[i * P + j];
        float u   = si + I[j * P + j] - in_;
        float iou = in_ / (u + 1e-6f);
        if (iou >= tau) { pick = j; break; }
    }
    for (int d = 0; d < 3; ++d)
        neg[(size_t)t * 3 + d] = (pick >= 0) ? pos[((size_t)b * P + pick) * 3 + d] : 0.f;
}

// =============================== host side ==================================
#define LAUNCH1D(kern, total, ...)                                                  \
    do { int _t = (int)(total); int _b = (_t + 255) / 256;                          \
         kern<<<_b, 256, 0, stream>>>(__VA_ARGS__); } while (0)

namespace {
struct Arena {
    char* base; size_t off;
    void* take(size_t bytes) {
        void* p = base + off;
        off = (off + bytes + 255) & ~(size_t)255;
        return p;
    }
};
struct BNL { const float *W, *b, *g, *bt, *mn, *vr; };
struct PrepL { _Float16* Wh; float* al; float* be; int Cout, Cin, Kpad; };
} // namespace

extern "C" void kernel_launch(void* const* d_in, const int* in_sizes, int n_in,
                              void* d_out, int out_size, void* d_ws, size_t ws_size,
                              hipStream_t stream)
{
    (void)in_sizes; (void)n_in; (void)out_size; (void)ws_size;
    const int B = 4, N = 8192, P = 8, D = 128;

    // ---- unpack inputs (setup_inputs dict insertion order) ----
    int ci = 0;
    const float* points     = (const float*)d_in[ci++];
    const float* pos_coords = (const float*)d_in[ci++];
    auto nextBN = [&]() {
        BNL l;
        l.W  = (const float*)d_in[ci++]; l.b  = (const float*)d_in[ci++];
        l.g  = (const float*)d_in[ci++]; l.bt = (const float*)d_in[ci++];
        l.mn = (const float*)d_in[ci++]; l.vr = (const float*)d_in[ci++];
        return l;
    };
    BNL sa1L[3], sa2L[3], sa3L[3], fp3L[2], fp2L[2], fp1L[3];
    for (int k = 0; k < 3; ++k) sa1L[k] = nextBN();
    for (int k = 0; k < 3; ++k) sa2L[k] = nextBN();
    for (int k = 0; k < 3; ++k) sa3L[k] = nextBN();
    for (int k = 0; k < 2; ++k) fp3L[k] = nextBN();
    for (int k = 0; k < 2; ++k) fp2L[k] = nextBN();
    for (int k = 0; k < 3; ++k) fp1L[k] = nextBN();
    const float* promptW = (const float*)d_in[ci++];
    const float* promptb = (const float*)d_in[ci++];
    const float* md0W = (const float*)d_in[ci++];
    const float* md0b = (const float*)d_in[ci++];
    const float* md1W = (const float*)d_in[ci++];
    const float* md1b = (const float*)d_in[ci++];
    const float* rk0W = (const float*)d_in[ci++];
    const float* rk0b = (const float*)d_in[ci++];
    const float* rk1W = (const float*)d_in[ci++];
    const float* rk1b = (const float*)d_in[ci++];

    // ---- output slices ----
    float* out_refined = (float*)d_out;                          // B*P*N
    float* out_feats   = out_refined + (size_t)B * P * N;        // B*N*D
    float* out_risk    = out_feats   + (size_t)B * N * D;        // B*N

    Arena A{(char*)d_ws, 0};

    // ---- fold BN + cast weights to padded f16 (persistent) ----
    auto prep = [&](BNL l, int Cout, int Cin) {
        PrepL p;
        p.Cout = Cout; p.Cin = Cin; p.Kpad = (Cin + 31) / 32 * 32;
        p.Wh = (_Float16*)A.take((size_t)Cout * p.Kpad * sizeof(_Float16));
        p.al = (float*)A.take((size_t)Cout * sizeof(float));
        p.be = (float*)A.take((size_t)Cout * sizeof(float));
        LAUNCH1D(prep_layer, Cout * p.Kpad, l.W, l.b, l.g, l.bt, l.mn, l.vr,
                 p.Wh, p.al, p.be, Cout, Cin, p.Kpad);
        return p;
    };
    PrepL Psa1[3] = { prep(sa1L[0],  64,   6), prep(sa1L[1],  64,  64), prep(sa1L[2], 128,  64) };
    PrepL Psa2[3] = { prep(sa2L[0], 128, 131), prep(sa2L[1], 128, 128), prep(sa2L[2], 256, 128) };
    PrepL Psa3[3] = { prep(sa3L[0], 256, 259), prep(sa3L[1], 512, 256), prep(sa3L[2],1024, 512) };
    PrepL Pfp3[2] = { prep(fp3L[0], 256,1280), prep(fp3L[1], 256, 256) };
    PrepL Pfp2[2] = { prep(fp2L[0], 256, 384), prep(fp2L[1], 128, 256) };
    PrepL Pfp1[3] = { prep(fp1L[0], 128, 131), prep(fp1L[1], 128, 128), prep(fp1L[2], 128, 128) };
    BNL md0 = { md0W, md0b, nullptr, nullptr, nullptr, nullptr };
    BNL rk0 = { rk0W, rk0b, nullptr, nullptr, nullptr, nullptr };
    PrepL Pmd0 = prep(md0, 64, 128);
    PrepL Prk0 = prep(rk0, 64, 128);

    // ---- persistent intermediates ----
    float* l1x    = (float*)A.take((size_t)B * 512 * 3 * 4);
    float* l2x    = (float*)A.take((size_t)B * 128 * 3 * 4);
    int*   idx1   = (int*)  A.take((size_t)B * 512 * 32 * 4);
    int*   idx2   = (int*)  A.take((size_t)B * 128 * 64 * 4);
    float* l1p    = (float*)A.take((size_t)B * 512 * 128 * 4);
    float* l2p    = (float*)A.take((size_t)B * 128 * 256 * 4);
    float* l3p    = (float*)A.take((size_t)B * 1024 * 4);
    float* fp3out = (float*)A.take((size_t)B * 128 * 256 * 4);
    float* fp2out = (float*)A.take((size_t)B * 512 * 128 * 4);
    float* pos_e  = (float*)A.take((size_t)B * P * D * 4);
    float* neg    = (float*)A.take((size_t)B * P * 3 * 4);
    float* neg_e  = (float*)A.take((size_t)B * P * D * 4);
    float* inter  = (float*)A.take((size_t)B * P * P * 4);
    float* logits = (float*)A.take((size_t)B * P * N * 4);
    const size_t scratch_base = A.off;

    // 64x64 macro-tiles per block (4 waves, 32x32 each)
    auto gemm = [&](const _Float16* X, const PrepL& L, _Float16* Yh, float* Yf, int M) {
        dim3 g(M / 64, L.Cout / 64);
        gemm_bn_relu_wmma<<<g, 128, 0, stream>>>(X, L.Wh, L.al, L.be, Yh, Yf,
                                                 M, L.Kpad, L.Cout);
    };

    // ---- centroids: l1x = points[:, :512], l2x = points[:, :128] ----
    LAUNCH1D(copy_first, B * 512 * 3, points, l1x, B, N, 512);
    LAUNCH1D(copy_first, B * 128 * 3, points, l2x, B, N, 128);

    // =========================== SA1 ===========================
    knn_kernel<<<B * 512, 256, (N + 512) * 4, stream>>>(l1x, points, idx1, 512, N, 32);
    {
        A.off = scratch_base;
        int M = B * 512 * 32;   // 65536
        _Float16* X0 = (_Float16*)A.take((size_t)M * 32 * 2);
        _Float16* A1 = (_Float16*)A.take((size_t)M * 64 * 2);
        _Float16* A2 = (_Float16*)A.take((size_t)M * 64 * 2);
        _Float16* A3 = (_Float16*)A.take((size_t)M * 128 * 2);
        LAUNCH1D(group_points, M, points, l1x, points, idx1, X0, B, 512, 32, N, 3, 32);
        gemm(X0, Psa1[0], A1, nullptr, M);
        gemm(A1, Psa1[1], A2, nullptr, M);
        gemm(A2, Psa1[2], A3, nullptr, M);
        LAUNCH1D(maxpool_ns, B * 512 * 128, A3, l1p, B * 512, 32, 128);
    }

    // =========================== SA2 ===========================
    knn_kernel<<<B * 128, 256, (512 + 512) * 4, stream>>>(l2x, l1x, idx2, 128, 512, 64);
    {
        A.off = scratch_base;
        int M = B * 128 * 64;   // 32768
        _Float16* X0 = (_Float16*)A.take((size_t)M * 160 * 2);
        _Float16* A1 = (_Float16*)A.take((size_t)M * 128 * 2);
        _Float16* A2 = (_Float16*)A.take((size_t)M * 128 * 2);
        _Float16* A3 = (_Float16*)A.take((size_t)M * 256 * 2);
        LAUNCH1D(group_points, M, l1x, l2x, l1p, idx2, X0, B, 128, 64, 512, 128, 160);
        gemm(X0, Psa2[0], A1, nullptr, M);
        gemm(A1, Psa2[1], A2, nullptr, M);
        gemm(A2, Psa2[2], A3, nullptr, M);
        LAUNCH1D(maxpool_ns, B * 128 * 256, A3, l2p, B * 128, 64, 256);
    }

    // =========================== SA3 (group-all) ===========================
    {
        A.off = scratch_base;
        int M = B * 128;        // 512
        _Float16* X0 = (_Float16*)A.take((size_t)M * 288 * 2);
        _Float16* A1 = (_Float16*)A.take((size_t)M * 256 * 2);
        _Float16* A2 = (_Float16*)A.take((size_t)M * 512 * 2);
        _Float16* A3 = (_Float16*)A.take((size_t)M * 1024 * 2);
        LAUNCH1D(group_all, M, l2x, l2p, X0, B, 128, 256, 288);
        gemm(X0, Psa3[0], A1, nullptr, M);
        gemm(A1, Psa3[1], A2, nullptr, M);
        gemm(A2, Psa3[2], A3, nullptr, M);
        LAUNCH1D(maxpool_ns, B * 1024, A3, l3p, B, 128, 1024);
    }

    // =========================== FP3 (S==1 broadcast) ===========================
    {
        A.off = scratch_base;
        int M = B * 128;        // 512
        _Float16* X0 = (_Float16*)A.take((size_t)M * 1280 * 2);
        _Float16* A1 = (_Float16*)A.take((size_t)M * 256 * 2);
        _Float16* A2 = (_Float16*)A.take((size_t)M * 256 * 2);
        LAUNCH1D(fp_build, M, l2x, (const float*)nullptr, l2p, l3p, X0,
                 B, 128, 1, 256, 1024, 1280);
        gemm(X0, Pfp3[0], A1, nullptr, M);
        gemm(A1, Pfp3[1], A2, fp3out, M);
    }

    // =========================== FP2 ===========================
    {
        A.off = scratch_base;
        int M = B * 512;        // 2048
        _Float16* X0 = (_Float16*)A.take((size_t)M * 384 * 2);
        _Float16* A1 = (_Float16*)A.take((size_t)M * 256 * 2);
        _Float16* A2 = (_Float16*)A.take((size_t)M * 128 * 2);
        LAUNCH1D(fp_build, M, l1x, l2x, l1p, fp3out, X0, B, 512, 128, 128, 256, 384);
        gemm(X0, Pfp2[0], A1, nullptr, M);
        gemm(A1, Pfp2[1], A2, fp2out, M);
    }

    // =========================== FP1 -> feats (d_out) ===========================
    {
        A.off = scratch_base;
        int M = B * N;          // 32768
        _Float16* X0 = (_Float16*)A.take((size_t)M * 160 * 2);
        _Float16* A1 = (_Float16*)A.take((size_t)M * 128 * 2);
        _Float16* A2 = (_Float16*)A.take((size_t)M * 128 * 2);
        _Float16* A3 = (_Float16*)A.take((size_t)M * 128 * 2);
        LAUNCH1D(fp_build, M, points, l1x, points, fp2out, X0, B, N, 512, 3, 128, 160);
        gemm(X0, Pfp1[0], A1, nullptr, M);
        gemm(A1, Pfp1[1], A2, nullptr, M);
        gemm(A2, Pfp1[2], A3, out_feats, M);
    }

    // =========================== prompt + mask head (pass 1) ===========================
    LAUNCH1D(linear3, B * P * D, pos_coords, promptW, promptb, pos_e, B * P, D);
    {
        A.off = scratch_base;
        int M = B * P * N;      // 262144
        _Float16* Xh = (_Float16*)A.take((size_t)M * 128 * 2);
        _Float16* Hh = (_Float16*)A.take((size_t)M * 64 * 2);
        LAUNCH1D(head_build, (long long)M * 128, out_feats, pos_e,
                 (const float*)nullptr, Xh, B, P, N, D);
        gemm(Xh, Pmd0, Hh, nullptr, M);
        LAUNCH1D(dense_final, M, Hh, md1W, md1b, logits, M, 64, 0);

        // ---- NPC: pick negative prompt coords ----
        npc_inter<<<B * P * P, 256, 0, stream>>>(logits, inter, B, P, N);
        LAUNCH1D(npc_select, B * P, inter, pos_coords, neg, B, P, 0.1f);
        LAUNCH1D(linear3, B * P * D, neg, promptW, promptb, neg_e, B * P, D);

        // ---- mask head (pass 2, refined) ----
        LAUNCH1D(head_build, (long long)M * 128, out_feats, pos_e, neg_e, Xh, B, P, N, D);
        gemm(Xh, Pmd0, Hh, nullptr, M);
        LAUNCH1D(dense_final, M, Hh, md1W, md1b, out_refined, M, 64, 0);
    }

    // =========================== risk head ===========================
    {
        A.off = scratch_base;
        int M = B * N;          // 32768
        _Float16* Xh = (_Float16*)A.take((size_t)M * 128 * 2);
        _Float16* Hh = (_Float16*)A.take((size_t)M * 64 * 2);
        LAUNCH1D(cast_pad, M * 128, out_feats, Xh, M, 128, 128);
        gemm(Xh, Prk0, Hh, nullptr, M);
        LAUNCH1D(dense_final, M, Hh, rk1W, rk1b, out_risk, M, 64, 1);
    }
}
